// SimpleTemporalGCN_79482664780415
// MI455X (gfx1250) — matrix operand order repaired and checked by reading, not compiled, not run on hardware
//
#include <hip/hip_runtime.h>
#include <math.h>

typedef __bf16 bf16;
typedef __attribute__((ext_vector_type(16))) __bf16 v16bf;
typedef __attribute__((ext_vector_type(8)))  float  v8f;

#define NN   100   // nodes
#define HD   256   // hidden
#define TT   128   // time dim
#define MS   136   // AnT LDS row stride (bf16): 68 dwords == 4 mod 64 -> conflict free
#define HS   264   // H   LDS row stride (bf16): 132 dwords == 4 mod 64
#define GS   136   // G^T LDS row stride (bf16)
#define KPAD 128   // padded K for the 100/112-sized contraction dims

static __device__ __forceinline__ unsigned pkbf(float a, float b) {
  union { bf16 h[2]; unsigned u; } t;
  t.h[0] = (bf16)a;
  t.h[1] = (bf16)b;
  return t.u;   // v_cvt_pk_bf16_f32
}

// ---- two 16x16 output tiles (shared A fragment), K-loop of WMMA ------------
// A: row-major [m][k] (bf16, stride As). B: stored transposed [n][k] (stride Bs).
// A frag (16x32 bf16): lane l (m=l&15), vgpr v holds k = (v&3)*2 + (l>>4)*8 + (v>>2)*16 (+1)
// B frag (32x16 bf16): lane l (n=l&15), vgpr v holds k = (v&3)*2 + (l>>4)*8 + (v>>2)*16 (+1)
static __device__ __forceinline__ void tile_mm2(const bf16* __restrict__ A, int As,
                                                const bf16* __restrict__ B, int Bs,
                                                int K, int mt, int nt0, int lane,
                                                v8f& acc0, v8f& acc1) {
  const int hi = lane >> 4;
  const int lr = lane & 15;
  const bf16* arow  = A + (mt * 16 + lr) * As;
  const bf16* brow0 = B + (nt0 * 16 + lr) * Bs;
  const bf16* brow1 = brow0 + 16 * Bs;
  for (int kb = 0; kb < K; kb += 32) {
    union { v16bf v; unsigned u[8]; } fa, fb0, fb1;
#pragma unroll
    for (int v = 0; v < 8; ++v) {
      const int ka = kb + ((v & 3) << 1) + (hi << 3) + ((v >> 2) << 4);
      const int kn = kb + ((v & 3) << 1) + (hi << 3) + ((v >> 2) << 4);
      fa.u[v]  = *(const unsigned*)(arow + ka);   // dword = {bf16[k], bf16[k+1]}
      fb0.u[v] = *(const unsigned*)(brow0 + kn);
      fb1.u[v] = *(const unsigned*)(brow1 + kn);
    }
    acc0 = __builtin_amdgcn_wmma_f32_16x16x32_bf16(false, fa.v, false, fb0.v,
                                                   (short)0, acc0, false, false);
    acc1 = __builtin_amdgcn_wmma_f32_16x16x32_bf16(false, fa.v, false, fb1.v,
                                                   (short)0, acc1, false, false);
  }
}

// H epilogue: 8 values walk down rows (same column n) -> 16-bit stores
static __device__ __forceinline__ void store_h(bf16* __restrict__ Hl, const v8f& acc,
                                               float bn, int mt, int n, int hi,
                                               bool relu) {
  bf16* base = Hl + (mt * 16 + 8 * hi) * HS + n;
#pragma unroll
  for (int v = 0; v < 8; ++v) {
    float x = acc[v] + bn;
    if (relu) x = fmaxf(x, 0.0f);
    base[v * HS] = (bf16)x;
  }
}

// G^T epilogue: 8 values are consecutive m within one row -> packed b128 store
static __device__ __forceinline__ void store_gt(bf16* __restrict__ Gt, const v8f& acc,
                                                int mt, int n, int hi) {
  unsigned* dst = (unsigned*)(Gt + n * GS + mt * 16 + 8 * hi);  // 16B aligned
  dst[0] = pkbf(acc[0], acc[1]);
  dst[1] = pkbf(acc[2], acc[3]);
  dst[2] = pkbf(acc[4], acc[5]);
  dst[3] = pkbf(acc[6], acc[7]);
}

// ---- prep kernel: batch-independent precompute -----------------------------
__global__ void __launch_bounds__(256)
gcn_prep(const float* __restrict__ time_, const float* __restrict__ W1,
         const float* __restrict__ W2, const float* __restrict__ W3,
         const float* __restrict__ te_w1, const float* __restrict__ te_b1,
         const float* __restrict__ te_w2, const float* __restrict__ te_b2,
         const float* __restrict__ e0_w, const float* __restrict__ e0_b,
         const float* __restrict__ e1_w, const float* __restrict__ e1_b,
         float* __restrict__ wsf, bf16* __restrict__ w1t,
         bf16* __restrict__ w2t, bf16* __restrict__ w3t) {
  const int tid = threadIdx.x;
  if (blockIdx.x == 0) {
    {
      float s1 = 0.f, s2 = 0.f;
      for (int h = 0; h < HD; ++h) {
        const float v = e1_w[h];
        s1 += e0_w[tid * HD + h] * v;
        s2 += e0_w[(HD + tid) * HD + h] * v;
      }
      wsf[tid] = s1;
      wsf[HD + tid] = s2;
    }
    __shared__ float wv[TT];
    __shared__ float c0s;
    if (tid < TT) {
      float s = 0.f;
      for (int t2 = 0; t2 < TT; ++t2) s += te_w2[tid * TT + t2] * e1_w[HD + t2];
      wv[tid] = s;
    }
    if (tid == 0) {
      float c = e1_b[0];
      for (int h = 0; h < HD; ++h) c += e0_b[h] * e1_w[h];
      for (int t2 = 0; t2 < TT; ++t2) c += te_b2[t2] * e1_w[HD + t2];
      c0s = c;
    }
    __syncthreads();
    if (tid < 32) {
      const float tb = time_[tid];
      float s = c0s;
      for (int t = 0; t < TT; ++t) {
        const float x = tb * te_w1[t] + te_b1[t];
        const float g = 0.5f * x * (1.0f + erff(x * 0.70710678118654752f));
        s += g * wv[t];
      }
      wsf[2 * HD + tid] = s;
    }
  } else {
    const int gt = (blockIdx.x - 1) * blockDim.x + tid;
    const int gs = (gridDim.x - 1) * blockDim.x;
    for (int idx = gt; idx < HD * KPAD; idx += gs) {   // W1^T, K padded 100->128
      const int n = idx >> 7, k = idx & 127;
      w1t[idx] = (k < NN) ? (bf16)W1[k * HD + n] : (bf16)0.0f;
    }
    for (int idx = gt; idx < HD * HD; idx += gs) {     // W2^T, W3^T
      const int n = idx >> 8, k = idx & 255;
      w2t[idx] = (bf16)W2[k * HD + n];
      w3t[idx] = (bf16)W3[k * HD + n];
    }
  }
}

// ---- main fused kernel: one workgroup per batch element --------------------
__global__ void __launch_bounds__(256)
gcn_batch(const float* __restrict__ X, const float* __restrict__ b1,
          const float* __restrict__ b2, const float* __restrict__ b3,
          const bf16* __restrict__ w1t, const bf16* __restrict__ w2t,
          const bf16* __restrict__ w3t, const float* __restrict__ wsf,
          float* __restrict__ out) {
  extern __shared__ char smem[];
  bf16* Ml = (bf16*)smem;             // AnT, 112 x MS
  bf16* Hl = Ml + 112 * MS;           // H,   112 x HS
  bf16* Gt = Hl + 112 * HS;           // G^T, 256 x GS
  float* dis = (float*)(Gt + 256 * GS);   // 112
  float* pq  = dis + 112;                 // p:[0..111], q:[112..223]

  const int b    = blockIdx.x;
  const int tid  = threadIdx.x;
  const int lane = tid & 31;
  const int wave = tid >> 5;
  const float* Xb = X + b * (NN * NN);

  // ---- degrees / normalization (self-loops force deg >= 1) ----
  if (tid < 112) {
    if (tid < NN) {
      float d = 0.f;
      for (int j = 0; j < NN; ++j) {
        const float x = Xb[j * NN + tid];
        d += (j == tid) ? 1.0f : ((x != 0.0f) ? 1.0f : 0.0f);
      }
      dis[tid] = rsqrtf(d);
    } else {
      dis[tid] = 0.f;
    }
  }
  __syncthreads();

  // ---- M = normalized A^T in bf16, zero-padded to 112 x 128 (+row pad) ----
  for (int idx = tid; idx < 112 * MS; idx += 256) {
    const int i = idx / MS, j = idx - i * MS;
    float val = 0.f;
    if (i < NN && j < NN) {
      const float ah = (i == j) ? 1.0f : ((Xb[j * NN + i] != 0.0f) ? 1.0f : 0.0f);
      val = dis[i] * dis[j] * ah;
    }
    Ml[idx] = (bf16)val;
  }
  // zero G^T padding columns k in [112,136) once (rows 112..127 of logical G)
  for (int idx = tid; idx < 256 * 24; idx += 256) {
    const int n = idx / 24, k = 112 + idx % 24;
    Gt[n * GS + k] = (bf16)0.0f;
  }
  __syncthreads();

  const v8f zero = {0.f, 0.f, 0.f, 0.f, 0.f, 0.f, 0.f, 0.f};
  const int lr = lane & 15, hi = lane >> 4;

  // ---- Stage 1: H = relu(M @ W1 + b1)   [K = 128], 1x2 n-blocking ----
  for (int g = wave; g < 56; g += 8) {
    const int mt = g % 7, nt0 = (g / 7) * 2;
    v8f a0 = zero, a1 = zero;
    tile_mm2(Ml, MS, w1t, KPAD, KPAD, mt, nt0, lane, a0, a1);
    const int n0 = nt0 * 16 + lr, n1 = n0 + 16;
    store_h(Hl, a0, b1[n0], mt, n0, hi, true);
    store_h(Hl, a1, b1[n1], mt, n1, hi, true);
  }
  __syncthreads();

  // ---- Stage 2: G^T = (H @ W2)^T   [K = 256] ----
  for (int g = wave; g < 56; g += 8) {
    const int mt = g % 7, nt0 = (g / 7) * 2;
    v8f a0 = zero, a1 = zero;
    tile_mm2(Hl, HS, w2t, HD, HD, mt, nt0, lane, a0, a1);
    const int n0 = nt0 * 16 + lr;
    store_gt(Gt, a0, mt, n0, hi);
    store_gt(Gt, a1, mt, n0 + 16, hi);
  }
  __syncthreads();

  // ---- Stage 3: H = relu(M @ G + b2)   [K = 128, B = G^T in LDS] ----
  for (int g = wave; g < 56; g += 8) {
    const int mt = g % 7, nt0 = (g / 7) * 2;
    v8f a0 = zero, a1 = zero;
    tile_mm2(Ml, MS, Gt, GS, KPAD, mt, nt0, lane, a0, a1);
    const int n0 = nt0 * 16 + lr, n1 = n0 + 16;
    store_h(Hl, a0, b2[n0], mt, n0, hi, true);
    store_h(Hl, a1, b2[n1], mt, n1, hi, true);
  }
  __syncthreads();

  // ---- Stage 4: G^T = (H @ W3)^T   [K = 256] ----
  for (int g = wave; g < 56; g += 8) {
    const int mt = g % 7, nt0 = (g / 7) * 2;
    v8f a0 = zero, a1 = zero;
    tile_mm2(Hl, HS, w3t, HD, HD, mt, nt0, lane, a0, a1);
    const int n0 = nt0 * 16 + lr;
    store_gt(Gt, a0, mt, n0, hi);
    store_gt(Gt, a1, mt, n0 + 16, hi);
  }
  __syncthreads();

  // ---- Stage 5: H = M @ G + b3 (no relu) ----
  for (int g = wave; g < 56; g += 8) {
    const int mt = g % 7, nt0 = (g / 7) * 2;
    v8f a0 = zero, a1 = zero;
    tile_mm2(Ml, MS, Gt, GS, KPAD, mt, nt0, lane, a0, a1);
    const int n0 = nt0 * 16 + lr, n1 = n0 + 16;
    store_h(Hl, a0, b3[n0], mt, n0, hi, false);
    store_h(Hl, a1, b3[n1], mt, n1, hi, false);
  }
  __syncthreads();

  // ---- factorized edge scoring: p = H@ev1, q = H@ev2 ----
  if (tid < NN) {
    float s = 0.f;
    for (int n = 0; n < HD; ++n) s += (float)Hl[tid * HS + n] * wsf[n];
    pq[tid] = s;
  } else if (tid >= 128 && tid < 128 + NN) {
    const int j = tid - 128;
    float s = 0.f;
    for (int n = 0; n < HD; ++n) s += (float)Hl[j * HS + n] * wsf[HD + n];
    pq[112 + j] = s;
  }
  __syncthreads();

  const float Sb = wsf[2 * HD + b];
  float* outb = out + b * NN * NN;
  for (int idx = tid; idx < NN * NN; idx += 256) {
    const int i = idx / NN, j = idx - i * NN;
    outb[idx] = pq[i] + pq[112 + j] + Sb;
  }
}

extern "C" void kernel_launch(void* const* d_in, const int* in_sizes, int n_in,
                              void* d_out, int out_size, void* d_ws, size_t ws_size,
                              hipStream_t stream) {
  const float* X     = (const float*)d_in[0];
  const float* time_ = (const float*)d_in[1];
  const float* W1    = (const float*)d_in[2];
  const float* b1    = (const float*)d_in[3];
  const float* W2    = (const float*)d_in[4];
  const float* b2    = (const float*)d_in[5];
  const float* W3    = (const float*)d_in[6];
  const float* b3    = (const float*)d_in[7];
  const float* te_w1 = (const float*)d_in[8];
  const float* te_b1 = (const float*)d_in[9];
  const float* te_w2 = (const float*)d_in[10];
  const float* te_b2 = (const float*)d_in[11];
  const float* e0_w  = (const float*)d_in[12];
  const float* e0_b  = (const float*)d_in[13];
  const float* e1_w  = (const float*)d_in[14];
  const float* e1_b  = (const float*)d_in[15];

  // workspace layout: f32 [ev1(256) | ev2(256) | S(32)] then bf16 W^T blocks
  float* wsf = (float*)d_ws;
  bf16* w1t = (bf16*)((char*)d_ws + 4096);                      // 256*128 bf16
  bf16* w2t = (bf16*)((char*)d_ws + 4096 + 65536);              // 256*256 bf16
  bf16* w3t = (bf16*)((char*)d_ws + 4096 + 65536 + 131072);     // 256*256 bf16

  gcn_prep<<<8, 256, 0, stream>>>(time_, W1, W2, W3, te_w1, te_b1, te_w2, te_b2,
                                  e0_w, e0_b, e1_w, e1_b, wsf, w1t, w2t, w3t);

  const size_t lds = (size_t)(112 * MS + 112 * HS + 256 * GS) * sizeof(bf16) +
                     (size_t)(112 + 224) * sizeof(float);       // 160,576 B
  gcn_batch<<<32, 256, lds, stream>>>(X, b1, b2, b3, w1t, w2t, w3t, wsf,
                                      (float*)d_out);
}
